// PointNet2BackboneCPU_23639499997598
// MI455X (gfx1250) — matrix operand order repaired and checked by reading, not compiled, not run on hardware
//
#include <hip/hip_runtime.h>
#include <hip/hip_bf16.h>

// ---------------------------------------------------------------------------
// PointNet++ backbone forward for gfx1250 (MI455X), wave32, WMMA f16->f32.
// GEMMs use v_wmma_f32_16x16x32_f16 with 4x2 register blocking per wave
// (64 output channels x 32 points), A/B fragments use contiguous k = 16*hi + s per lane half.
// ---------------------------------------------------------------------------

typedef __attribute__((ext_vector_type(16))) _Float16 v16h;
typedef __attribute__((ext_vector_type(8)))  _Float16 v8h;
typedef __attribute__((ext_vector_type(8)))  float    v8f;

#define BN_EPS 1e-5f
#define MT 4   // m-tiles (16 ch each) per wave
#define NT 2   // n-tiles (16 rows each) per wave

__device__ __forceinline__ int lin_idx(int j, int Nsrc, int Ndst) {
  // emulate jnp.linspace(0, Nsrc-1, Ndst).astype(int32)
  return (int)(((double)j * (double)(Nsrc - 1)) / (double)(Ndst - 1));
}

// ---------------- WMMA GEMM:  Y(rows x M) = X(rows x K) * A(M x K)^T --------
// A: weights f16, row-major (M x Kpad).  X: activations f16, point-major
// (rows x Kpad).  Y: f32 point-major (rows x M).  rows %32==0, M %64==0,
// K %32==0 (all padded).
__global__ void k_gemm_wmma(const _Float16* __restrict__ A,
                            const _Float16* __restrict__ X,
                            float* __restrict__ Y,
                            int rows, int M, int K)
{
  const int lane = threadIdx.x & 31;
  const int wave = threadIdx.x >> 5;
  const int nblk = blockIdx.x * 8 + wave;        // 32-row macro block
  if (nblk * 32 >= rows) return;                 // wave-uniform, EXEC stays full
  const int mbase = blockIdx.y * (MT * 16);
  const int l15 = lane & 15;
  const int hi  = (lane >> 4);                   // 0 for lanes 0-15, 1 for 16-31

  // A fragment: row m; contiguous K run at hi*16+{0..15}
  const _Float16* Ap = A + (size_t)(mbase + l15) * K + hi * 16;
  // B fragment: col n; contiguous K run at (lane&16)+{0..15}
  const _Float16* Xp = X + (size_t)(nblk * 32 + l15) * K + hi * 16;
  const size_t Astep = (size_t)16 * K;

  v8f acc[MT][NT] = {};
  for (int k = 0; k < K; k += 32) {
    // WGP-scope prefetch of the next X k-chunk while WMMAs retire
    __builtin_prefetch(Xp + k + 32, 0, 3);
    v16h a[MT], b[NT];
#pragma unroll
    for (int m = 0; m < MT; ++m) {
      v8h lo  = *(const v8h*)(Ap + (size_t)m * Astep + k);
      v8h hi8 = *(const v8h*)(Ap + (size_t)m * Astep + k + 8);
      a[m] = __builtin_shufflevector(lo, hi8,
                                     0, 1, 2, 3, 4, 5, 6, 7,
                                     8, 9, 10, 11, 12, 13, 14, 15);
    }
#pragma unroll
    for (int n = 0; n < NT; ++n)
      b[n] = *(const v16h*)(Xp + (size_t)n * Astep + k);
#pragma unroll
    for (int m = 0; m < MT; ++m)
#pragma unroll
      for (int n = 0; n < NT; ++n)
        acc[m][n] = __builtin_amdgcn_wmma_f32_16x16x32_f16(
            false, a[m], false, b[n], (short)0, acc[m][n], false, false);
  }
  // D: lane owns col n, rows m = mbase + m*16 + hi*8 + {0..7} (contiguous)
#pragma unroll
  for (int m = 0; m < MT; ++m)
#pragma unroll
    for (int n = 0; n < NT; ++n) {
      float* Yp = Y + (size_t)(nblk * 32 + n * 16 + l15) * M
                    + mbase + m * 16 + hi * 8;
      *(v8f*)Yp = acc[m][n];
    }
}

// ---------------- BatchNorm pieces -----------------------------------------
__global__ void k_bn_stats(const float* __restrict__ Y, int rows, int M,
                           int npad_mask, int nvalid,
                           float* __restrict__ sums, float* __restrict__ sumsq)
{
  int r0 = blockIdx.x * 32;
  for (int c = threadIdx.x; c < M; c += blockDim.x) {
    float s = 0.f, ss = 0.f;
    for (int i = 0; i < 32; ++i) {
      int r = r0 + i;
      if (r >= rows) break;
      if ((r & npad_mask) < nvalid) {
        float v = Y[(size_t)r * M + c];
        s += v; ss += v * v;
      }
    }
    atomicAdd(&sums[c], s);
    atomicAdd(&sumsq[c], ss);
  }
}

__global__ void k_bn_finalize(const float* __restrict__ sums,
                              const float* __restrict__ sumsq,
                              const float* __restrict__ g,
                              const float* __restrict__ beta,
                              float* __restrict__ scale,
                              float* __restrict__ shift,
                              int M, float inv_cnt)
{
  int c = blockIdx.x * blockDim.x + threadIdx.x;
  if (c >= M) return;
  float m = sums[c] * inv_cnt;
  float v = sumsq[c] * inv_cnt - m * m;
  float sc = g[c] * rsqrtf(v + BN_EPS);
  scale[c] = sc;
  shift[c] = beta[c] - m * sc;
}

__global__ void k_bn_apply(const float* __restrict__ Y,
                           const float* __restrict__ scale,
                           const float* __restrict__ shift,
                           _Float16* __restrict__ dst,
                           int M, int npad_mask, int nvalid)
{
  int r = blockIdx.x;
  bool ok = (r & npad_mask) < nvalid;
  for (int c = threadIdx.x; c < M; c += blockDim.x) {
    float v = 0.f;
    if (ok) {
      v = Y[(size_t)r * M + c] * scale[c] + shift[c];
      v = v > 0.f ? v : 0.f;
    }
    dst[(size_t)r * M + c] = (_Float16)v;
  }
}

// Final layer: write f32 channel-major (B, C, N) into d_out
__global__ void k_bn_apply_final(const float* __restrict__ Y,
                                 const float* __restrict__ scale,
                                 const float* __restrict__ shift,
                                 float* __restrict__ out,
                                 int Nn, int M)
{
  int r = blockIdx.x;                // r = b*Nn + n
  int b = r / Nn, n = r - b * Nn;
  for (int c = threadIdx.x; c < M; c += blockDim.x) {
    float v = Y[(size_t)r * M + c] * scale[c] + shift[c];
    v = v > 0.f ? v : 0.f;
    out[((size_t)b * M + c) * Nn + n] = v;
  }
}

// ---------------- data movement helpers ------------------------------------
__global__ void k_w2h(const float* __restrict__ W, _Float16* __restrict__ dst,
                      int M, int K, int Kpad)
{
  size_t i = blockIdx.x * (size_t)blockDim.x + threadIdx.x;
  if (i >= (size_t)M * Kpad) return;
  int m = (int)(i / Kpad), k = (int)(i % Kpad);
  dst[i] = (_Float16)(k < K ? W[(size_t)m * K + k] : 0.f);
}

__global__ void k_copy_f32(float* __restrict__ dst, const float* __restrict__ src, int n)
{
  int i = blockIdx.x * blockDim.x + threadIdx.x;
  if (i < n) dst[i] = src[i];
}

// features (B, Nsrc, 6) f32 -> X0 (B*Ndst, 32) f16, gathered by linspace
__global__ void k_build_x0(const float* __restrict__ feats, _Float16* __restrict__ dst,
                           int B, int Nsrc, int Ndst)
{
  int t = blockIdx.x * blockDim.x + threadIdx.x;
  if (t >= B * Ndst) return;
  int b = t / Ndst, j = t - b * Ndst;
  int idx = lin_idx(j, Nsrc, Ndst);
  const float* s = feats + ((size_t)b * Nsrc + idx) * 6;
  _Float16* d = dst + (size_t)t * 32;
  for (int c = 0; c < 32; ++c) d[c] = (_Float16)(c < 6 ? s[c] : 0.f);
}

// gather f16 rows by linspace, same channel count on both sides
__global__ void k_gather_lin_f16(const _Float16* __restrict__ src, _Float16* __restrict__ dst,
                                 int Nsrc, int Ndst, int C)
{
  int r = blockIdx.x;                 // r = b*Ndst + j
  int b = r / Ndst, j = r - b * Ndst;
  int idx = lin_idx(j, Nsrc, Ndst);
  const _Float16* s = src + ((size_t)b * Nsrc + idx) * C;
  _Float16* d = dst + (size_t)r * C;
  for (int c = threadIdx.x; c < C; c += blockDim.x) d[c] = s[c];
}

// gather xyz-like f32 rows (3 channels) by linspace
__global__ void k_gather_pts(const float* __restrict__ src, float* __restrict__ dst,
                             int B, int Nsrc, int Ndst)
{
  int t = blockIdx.x * blockDim.x + threadIdx.x;
  if (t >= B * Ndst) return;
  int b = t / Ndst, j = t - b * Ndst;
  int idx = lin_idx(j, Nsrc, Ndst);
  const float* s = src + ((size_t)b * Nsrc + idx) * 3;
  float* d = dst + (size_t)t * 3;
  d[0] = s[0]; d[1] = s[1]; d[2] = s[2];
}

// maxpool f2 (B*Np x C) over points -> X (B*16 x C), row 0 valid, rows 1..15 zero
__global__ void k_maxpool_build(const _Float16* __restrict__ f2, _Float16* __restrict__ dst,
                                int Np, int C)
{
  int b = blockIdx.x;
  for (int c = threadIdx.x; c < C; c += blockDim.x) {
    float m = -3.0e38f;
    for (int j = 0; j < Np; ++j) {
      float v = (float)f2[((size_t)b * Np + j) * C + c];
      m = v > m ? v : m;
    }
    dst[((size_t)b * 16) * C + c] = (_Float16)m;
    for (int j = 1; j < 16; ++j) dst[((size_t)b * 16 + j) * C + c] = (_Float16)0.f;
  }
}

// FP0 concat: broadcast global feature (1024) + f2 (256) -> (B*128 x 1280)
__global__ void k_concat_fp0(const _Float16* __restrict__ gfeat,
                             const _Float16* __restrict__ f2,
                             _Float16* __restrict__ dst, int Np)
{
  int r = blockIdx.x;                 // r = b*Np + j
  int b = r / Np;
  const _Float16* gp = gfeat + ((size_t)b * 16) * 1024;
  const _Float16* fp = f2 + (size_t)r * 256;
  _Float16* d = dst + (size_t)r * 1280;
  for (int c = threadIdx.x; c < 1280; c += blockDim.x)
    d[c] = (c < 1024) ? gp[c] : fp[c - 1024];
}

// nearest-neighbour index: for each target point, argmin over Ns source points
__global__ void k_nn_idx(const float* __restrict__ tgt, const float* __restrict__ src,
                         int* __restrict__ idxout, int Nt, int Ns)
{
  __shared__ float sp[512 * 3];
  int b = blockIdx.y;
  for (int i = threadIdx.x; i < Ns * 3; i += blockDim.x)
    sp[i] = src[(size_t)b * Ns * 3 + i];
  __syncthreads();
  int t = blockIdx.x * blockDim.x + threadIdx.x;
  if (t >= Nt) return;
  const float* p = tgt + ((size_t)b * Nt + t) * 3;
  float x = p[0], y = p[1], z = p[2];
  float best = 3.0e38f; int bi = 0;
  for (int s = 0; s < Ns; ++s) {
    float dx = x - sp[3 * s], dy = y - sp[3 * s + 1], dz = z - sp[3 * s + 2];
    float d = dx * dx + dy * dy + dz * dz;
    if (d < best) { best = d; bi = s; }
  }
  idxout[(size_t)b * Nt + t] = bi;
}

// concat: NN-gathered coarse features (f16) + skip features (f16)
__global__ void k_concat_nn(const _Float16* __restrict__ h, const _Float16* __restrict__ tf,
                            const int* __restrict__ nn, _Float16* __restrict__ dst,
                            int Nt, int Ns, int Ch, int Ctf)
{
  int r = blockIdx.x;                 // r = b*Nt + t
  int b = r / Nt;
  const _Float16* hp = h + ((size_t)b * Ns + nn[r]) * Ch;
  const _Float16* tp = tf + (size_t)r * Ctf;
  _Float16* d = dst + (size_t)r * (Ch + Ctf);
  int C = Ch + Ctf;
  for (int c = threadIdx.x; c < C; c += blockDim.x)
    d[c] = (c < Ch) ? hp[c] : tp[c - Ch];
}

// concat: NN-gathered h1 (f16,128) + raw features (f32,6), pad to Cdst
__global__ void k_concat_nn_f32(const _Float16* __restrict__ h, const float* __restrict__ feats,
                                const int* __restrict__ nn, _Float16* __restrict__ dst,
                                int Nt, int Ns, int Ch, int Craw, int Cdst)
{
  int r = blockIdx.x;
  int b = r / Nt;
  const _Float16* hp = h + ((size_t)b * Ns + nn[r]) * Ch;
  const float* tp = feats + (size_t)r * Craw;
  _Float16* d = dst + (size_t)r * Cdst;
  for (int c = threadIdx.x; c < Cdst; c += blockDim.x) {
    float v;
    if (c < Ch)            v = (float)hp[c];
    else if (c < Ch + Craw) v = tp[c - Ch];
    else                   v = 0.f;
    d[c] = (_Float16)v;
  }
}

// ---------------- host-side layer driver -----------------------------------
struct BnTmp { float *sums, *sumsq, *scale, *shift; };

static void run_layer(const float* Wf, const float* g, const float* beta,
                      _Float16* wbuf, const _Float16* Xin, float* Y, _Float16* Xout,
                      int rows, int M, int K, int Kpad,
                      int npad_mask, int nvalid, int Bcnt,
                      const BnTmp& t, hipStream_t stream)
{
  int wtot = M * Kpad;
  k_w2h<<<(wtot + 255) / 256, 256, 0, stream>>>(Wf, wbuf, M, K, Kpad);
  dim3 grid((rows / 32 + 7) / 8, M / 64);
  k_gemm_wmma<<<grid, 256, 0, stream>>>(wbuf, Xin, Y, rows, M, Kpad);
  hipMemsetAsync(t.sums, 0, M * sizeof(float), stream);
  hipMemsetAsync(t.sumsq, 0, M * sizeof(float), stream);
  k_bn_stats<<<(rows + 31) / 32, 256, 0, stream>>>(Y, rows, M, npad_mask, nvalid,
                                                   t.sums, t.sumsq);
  float inv_cnt = 1.0f / (float)(Bcnt * nvalid);
  k_bn_finalize<<<(M + 127) / 128, 128, 0, stream>>>(t.sums, t.sumsq, g, beta,
                                                     t.scale, t.shift, M, inv_cnt);
  if (Xout)
    k_bn_apply<<<rows, 128, 0, stream>>>(Y, t.scale, t.shift, Xout, M, npad_mask, nvalid);
}

extern "C" void kernel_launch(void* const* d_in, const int* in_sizes, int n_in,
                              void* d_out, int out_size, void* d_ws, size_t ws_size,
                              hipStream_t stream)
{
  (void)in_sizes; (void)n_in; (void)out_size; (void)ws_size;
  const int B = 8, N = 16384;
  const float* xyz   = (const float*)d_in[0];   // (8,16384,3)
  const float* feats = (const float*)d_in[1];   // (8,16384,6)

  // param leaf order (jax pytree, sorted keys): fp[0..2] then sa[0..2];
  // per layer: W, b, beta, g  (b unused: cancels in BN)
  const int FP00 = 2, FP01 = 6, FP10 = 10, FP11 = 14, FP20 = 18, FP21 = 22;
  const int SA00 = 26, SA01 = 30, SA02 = 34;
  const int SA10 = 38, SA11 = 42, SA12 = 46;
  const int SA20 = 50, SA21 = 54, SA22 = 58;
  auto Wp = [&](int i) { return (const float*)d_in[i]; };
  auto Bp = [&](int i) { return (const float*)d_in[i + 2]; };  // beta
  auto Gp = [&](int i) { return (const float*)d_in[i + 3]; };  // g

  // ---- workspace bump allocator (deterministic per call) ----
  char* base = (char*)d_ws;
  size_t off = 0;
  auto alloc = [&](size_t bytes) -> char* {
    char* p = base + off;
    off += (bytes + 255) & ~(size_t)255;
    return p;
  };
  // arenas
  _Float16* XA = (_Float16*)alloc((size_t)131072 * 160 * 2);
  _Float16* XB = (_Float16*)alloc((size_t)131072 * 160 * 2);
  float*    Yb = (float*)   alloc((size_t)131072 * 128 * 4);
  // persistent
  _Float16* F1 = (_Float16*)alloc((size_t)4096 * 128 * 2);   // sa_f[1]
  _Float16* F2 = (_Float16*)alloc((size_t)1024 * 256 * 2);   // sa_f[2]
  _Float16* GF = (_Float16*)alloc((size_t)128 * 1024 * 2);   // global feature
  _Float16* H0 = (_Float16*)alloc((size_t)1024 * 256 * 2);   // fp0 out
  _Float16* H1 = (_Float16*)alloc((size_t)4096 * 128 * 2);   // fp1 out
  float* PTS1 = (float*)alloc((size_t)4096 * 3 * 4);
  float* PTS2 = (float*)alloc((size_t)1024 * 3 * 4);
  int* NN1 = (int*)alloc((size_t)4096 * 4);
  int* NN2 = (int*)alloc((size_t)131072 * 4);
  BnTmp bt;
  bt.sums  = (float*)alloc(1024 * 4);
  bt.sumsq = (float*)alloc(1024 * 4);
  bt.scale = (float*)alloc(1024 * 4);
  bt.shift = (float*)alloc(1024 * 4);
  // f16 weight buffers, one per layer
  auto wa = [&](int M, int Kpad) { return (_Float16*)alloc((size_t)M * Kpad * 2); };
  _Float16 *w_sa00 = wa(64, 32),   *w_sa01 = wa(64, 64),   *w_sa02 = wa(128, 64);
  _Float16 *w_sa10 = wa(128, 128), *w_sa11 = wa(128, 128), *w_sa12 = wa(256, 128);
  _Float16 *w_sa20 = wa(256, 256), *w_sa21 = wa(512, 256), *w_sa22 = wa(1024, 512);
  _Float16 *w_fp00 = wa(512, 1280), *w_fp01 = wa(256, 512);
  _Float16 *w_fp10 = wa(256, 384),  *w_fp11 = wa(128, 256);
  _Float16 *w_fp20 = wa(128, 160),  *w_fp21 = wa(128, 128);

  // ---- points subsampling ----
  k_gather_pts<<<(B * 512 + 127) / 128, 128, 0, stream>>>(xyz, PTS1, B, N, 512);
  k_gather_pts<<<(B * 128 + 127) / 128, 128, 0, stream>>>(PTS1, PTS2, B, 512, 128);

  // ---- SA1: gather features (6ch, pad 32) at 512 pts; MLP 6->64->64->128 ----
  k_build_x0<<<(B * 512 + 127) / 128, 128, 0, stream>>>(feats, XA, B, N, 512);
  run_layer(Wp(SA00), Gp(SA00), Bp(SA00), w_sa00, XA, Yb, XB, 4096, 64, 6, 32, 511, 512, B, bt, stream);
  run_layer(Wp(SA01), Gp(SA01), Bp(SA01), w_sa01, XB, Yb, XA, 4096, 64, 64, 64, 511, 512, B, bt, stream);
  run_layer(Wp(SA02), Gp(SA02), Bp(SA02), w_sa02, XA, Yb, F1, 4096, 128, 64, 64, 511, 512, B, bt, stream);

  // ---- SA2: gather F1 to 128 pts; MLP 128->128->128->256 ----
  k_gather_lin_f16<<<B * 128, 128, 0, stream>>>(F1, XA, 512, 128, 128);
  run_layer(Wp(SA10), Gp(SA10), Bp(SA10), w_sa10, XA, Yb, XB, 1024, 128, 128, 128, 127, 128, B, bt, stream);
  run_layer(Wp(SA11), Gp(SA11), Bp(SA11), w_sa11, XB, Yb, XA, 1024, 128, 128, 128, 127, 128, B, bt, stream);
  run_layer(Wp(SA12), Gp(SA12), Bp(SA12), w_sa12, XA, Yb, F2, 1024, 256, 128, 128, 127, 128, B, bt, stream);

  // ---- SA3: maxpool F2 -> 1 point (padded to 16 rows); MLP 256->256->512->1024 ----
  k_maxpool_build<<<B, 128, 0, stream>>>(F2, XA, 128, 256);
  run_layer(Wp(SA20), Gp(SA20), Bp(SA20), w_sa20, XA, Yb, XB, 128, 256, 256, 256, 15, 1, B, bt, stream);
  run_layer(Wp(SA21), Gp(SA21), Bp(SA21), w_sa21, XB, Yb, XA, 128, 512, 256, 256, 15, 1, B, bt, stream);
  run_layer(Wp(SA22), Gp(SA22), Bp(SA22), w_sa22, XA, Yb, GF, 128, 1024, 512, 512, 15, 1, B, bt, stream);

  // ---- FP0: broadcast GF + F2 -> 1280ch @ 128 pts; MLP 1280->512->256 ----
  k_concat_fp0<<<B * 128, 256, 0, stream>>>(GF, F2, XA, 128);
  run_layer(Wp(FP00), Gp(FP00), Bp(FP00), w_fp00, XA, Yb, XB, 1024, 512, 1280, 1280, 127, 128, B, bt, stream);
  run_layer(Wp(FP01), Gp(FP01), Bp(FP01), w_fp01, XB, Yb, H0, 1024, 256, 512, 512, 127, 128, B, bt, stream);

  // ---- FP1: NN(512->128), concat H0-gather (256) + F1 (128); MLP 384->256->128 ----
  {
    dim3 g((512 + 255) / 256, B);
    k_nn_idx<<<g, 256, 0, stream>>>(PTS1, PTS2, NN1, 512, 128);
  }
  k_concat_nn<<<B * 512, 128, 0, stream>>>(H0, F1, NN1, XA, 512, 128, 256, 128);
  run_layer(Wp(FP10), Gp(FP10), Bp(FP10), w_fp10, XA, Yb, XB, 4096, 256, 384, 384, 511, 512, B, bt, stream);
  run_layer(Wp(FP11), Gp(FP11), Bp(FP11), w_fp11, XB, Yb, H1, 4096, 128, 256, 256, 511, 512, B, bt, stream);

  // ---- FP2: NN(16384->512), concat H1-gather (128) + raw feats (6) pad 160;
  //           MLP 134->128->128; final layer writes d_out channel-major ----
  {
    dim3 g((16384 + 255) / 256, B);
    k_nn_idx<<<g, 256, 0, stream>>>(xyz, PTS1, NN2, 16384, 512);
  }
  k_concat_nn_f32<<<B * 16384, 128, 0, stream>>>(H1, feats, NN2, XA, 16384, 512, 128, 6, 160);
  run_layer(Wp(FP20), Gp(FP20), Bp(FP20), w_fp20, XA, Yb, XB, 131072, 128, 134, 160, 16383, 16384, B, bt, stream);
  // last layer: GEMM + BN stats, then apply directly into d_out (B, 128, N) f32
  run_layer(Wp(FP21), Gp(FP21), Bp(FP21), w_fp21, XB, Yb, nullptr, 131072, 128, 128, 128, 16383, 16384, B, bt, stream);
  float* out = (float*)d_out;
  k_copy_f32<<<(B * N * 3 + 255) / 256, 256, 0, stream>>>(out, xyz, B * N * 3);
  k_bn_apply_final<<<131072, 128, 0, stream>>>(Yb, bt.scale, bt.shift,
                                               out + (size_t)B * N * 3, N, 128);
}
